// Head_42348377539002
// MI455X (gfx1250) — matrix operand order repaired and checked by reading, not compiled, not run on hardware
//
#include <hip/hip_runtime.h>
#include <stdint.h>

// Problem constants (from reference): B=128, T=256, C=384, H=6, d=64
#define Bb 128
#define Tt 256
#define Cc 384
#define Hh 6
#define Dd 64
#define BT (Bb * Tt)

// LDS strides (elements); all row pitches are multiples of 8 elems (16B)
#define XW_LD 40
#define QL_LD 72
#define KL_LD 72
#define VT_LD 264
#define PL_LD 264
#define SL_LD 260

typedef __attribute__((ext_vector_type(16))) __bf16 v16bf;
typedef __attribute__((ext_vector_type(8)))  float  v8f;

// ---------------------------------------------------------------------------
// CDNA5 async global->LDS copy (ASYNCcnt-tracked) with vectorized fallback.
// Probe-confirmed signature: (v4i addrspace(1)*, v4i addrspace(3)*, Ii, Ii)
// ---------------------------------------------------------------------------
#if defined(__gfx1250__) &&                                         \
    __has_builtin(__builtin_amdgcn_global_load_async_to_lds_b128) && \
    __has_builtin(__builtin_amdgcn_s_wait_asynccnt)
#define HAVE_ASYNC_LDS 1
#endif

#ifdef HAVE_ASYNC_LDS
typedef int v4i_vs __attribute__((vector_size(16)));
typedef __attribute__((address_space(1))) v4i_vs* as1_v4i_p;
typedef __attribute__((address_space(3))) v4i_vs* as3_v4i_p;
#endif

__device__ __forceinline__ void async_copy16(const void* gsrc, void* ldst) {
#ifdef HAVE_ASYNC_LDS
  as1_v4i_p g = (as1_v4i_p)(unsigned long long)gsrc;
  // generic LDS pointer = {shared aperture, offset}; low 32 bits are the
  // LDS byte offset, which is what an AS(3) pointer holds.
  as3_v4i_p l = (as3_v4i_p)(unsigned int)(unsigned long long)ldst;
  __builtin_amdgcn_global_load_async_to_lds_b128(g, l, 0, 0);
#else
  *(uint4*)ldst = *(const uint4*)gsrc;
#endif
}

__device__ __forceinline__ void async_copy_fence() {
#ifdef HAVE_ASYNC_LDS
  __builtin_amdgcn_s_wait_asynccnt(0);
#endif
}

// ---------------------------------------------------------------------------
// WMMA operand loaders (CDNA5 wave32 layouts, CDNA5 ISA §7.12.2)
// A (16x32 bf16, MxK): lanes 0-15 -> M=lane, K 0..7 / 16..23;
//                      lanes 16-31 -> M=lane-16, K 8..15 / 24..31.
// B (32x16 bf16, KxN): stored transposed in LDS as Bt[n][k];
//                      lanes 0-15 -> N=lane, K 0..7 / 16..23; lanes 16-31 -> K 8..15 / 24..31.
// C/D (16x16 f32):     lanes 0-15 -> N=lane, VGPR r = M=r; lanes 16-31 M=8+r.
// ---------------------------------------------------------------------------
__device__ __forceinline__ v16bf load_a16(const __bf16* base, int ld, int lane) {
  int m  = lane & 15;
  int kb = (lane >> 4) << 3;            // 0 or 8
  const __bf16* row = base + m * ld;
  union { v16bf v; uint4 q[2]; } u;
  u.q[0] = *(const uint4*)(row + kb);        // K = kb .. kb+7
  u.q[1] = *(const uint4*)(row + 16 + kb);   // K = 16+kb .. 16+kb+7
  return u.v;
}

__device__ __forceinline__ v16bf load_bt16(const __bf16* base, int ld, int lane) {
  int n  = lane & 15;
  int kb = (lane >> 4) << 3;            // 0 or 8
  const __bf16* row = base + n * ld;
  union { v16bf v; uint4 q[2]; } u;
  u.q[0] = *(const uint4*)(row + kb);        // K = kb .. kb+7
  u.q[1] = *(const uint4*)(row + 16 + kb);   // K = 16+kb .. 16+kb+7
  return u.v;
}

__device__ __forceinline__ v8f wmma_bf16(v16bf a, v16bf b, v8f c) {
  return __builtin_amdgcn_wmma_f32_16x16x32_bf16(
      /*neg_a=*/false, a, /*neg_b=*/false, b,
      /*c_mod=*/(short)0, c, /*reuse_a=*/false, /*reuse_b=*/false);
}

// ---------------------------------------------------------------------------
// Kernel 0: f32 -> bf16 conversion, 4 elements/thread (b128 in, b64 out)
// ---------------------------------------------------------------------------
__global__ void cvt_bf16_kernel(const float4* __restrict__ src,
                                uint2* __restrict__ dst, int n4) {
  int i = blockIdx.x * blockDim.x + threadIdx.x;
  if (i < n4) {
    float4 f = src[i];
    union { __bf16 h[4]; uint2 u; } pk;
    pk.h[0] = (__bf16)f.x; pk.h[1] = (__bf16)f.y;
    pk.h[2] = (__bf16)f.z; pk.h[3] = (__bf16)f.w;
    dst[i] = pk.u;
  }
}

// ---------------------------------------------------------------------------
// Kernel 1: QKV projection GEMM, double-buffered K-loop.
// Grid: (BT/64, C/64, 3). Block: 256 threads = 8 waves.
// X tile async-copied to LDS; W tile loaded b128 + stored transposed.
// ---------------------------------------------------------------------------
__launch_bounds__(256, 2)
__global__ void qkv_gemm_kernel(const __bf16* __restrict__ xb,
                                const __bf16* __restrict__ w0,
                                const __bf16* __restrict__ w1,
                                const __bf16* __restrict__ w2,
                                __bf16* __restrict__ qo,
                                __bf16* __restrict__ ko,
                                __bf16* __restrict__ vo) {
  __shared__ __bf16 Xl[2][64 * XW_LD];   // [m][k], k padded 32->40 (row 80B)
  __shared__ __bf16 Wl[2][64 * XW_LD];   // transposed: [n][k]

  const int tid  = threadIdx.x;
  const int lane = tid & 31;
  const int w    = tid >> 5;
  const int mi   = w & 3;
  const int ni0  = (w >> 2) << 1;
  const int m0   = blockIdx.x * 64;
  const int n0   = blockIdx.y * 64;

  const __bf16* W   = (blockIdx.z == 0) ? w0 : (blockIdx.z == 1 ? w1 : w2);
  __bf16*       Out = (blockIdx.z == 0) ? qo : (blockIdx.z == 1 ? ko : vo);

  // per-thread 16B chunks: X tile 64 rows x 4 chunks, W tile 32 rows x 8 chunks
  const int xr = tid >> 2, xc = (tid & 3) * 8;
  const int wk = tid >> 3, wn = (tid & 7) * 8;

  // prologue: stage kt=0
  async_copy16(xb + (size_t)(m0 + xr) * Cc + xc, &Xl[0][xr * XW_LD + xc]);
  union { uint4 q; __bf16 h[8]; } wreg;
  wreg.q = *(const uint4*)(W + (size_t)wk * Cc + n0 + wn);

  v8f acc0 = {0.f, 0.f, 0.f, 0.f, 0.f, 0.f, 0.f, 0.f};
  v8f acc1 = acc0;

  for (int kt = 0; kt < Cc / 32; ++kt) {
    const int cur = kt & 1;
    // finish staging buffer `cur`: scatter W regs transposed into LDS
#pragma unroll
    for (int j = 0; j < 8; ++j) Wl[cur][(wn + j) * XW_LD + wk] = wreg.h[j];
    async_copy_fence();          // X(cur) async writes complete
    __syncthreads();             // publish buffer `cur`

    // prefetch stage kt+1 into the other buffer while computing on `cur`
    if (kt + 1 < Cc / 32) {
      int k0n = (kt + 1) * 32;
      async_copy16(xb + (size_t)(m0 + xr) * Cc + k0n + xc,
                   &Xl[cur ^ 1][xr * XW_LD + xc]);
      wreg.q = *(const uint4*)(W + (size_t)(k0n + wk) * Cc + n0 + wn);
    }

    v16bf a  = load_a16(&Xl[cur][mi * 16 * XW_LD], XW_LD, lane);
    v16bf b0 = load_bt16(&Wl[cur][ni0 * 16 * XW_LD], XW_LD, lane);
    v16bf b1 = load_bt16(&Wl[cur][(ni0 + 1) * 16 * XW_LD], XW_LD, lane);
    acc0 = wmma_bf16(a, b0, acc0);
    acc1 = wmma_bf16(a, b1, acc1);
  }

  const int n = lane & 15, mb = (lane >> 4) << 3;
#pragma unroll
  for (int r = 0; r < 8; ++r) {
    int mrow = m0 + mi * 16 + mb + r;
    Out[(size_t)mrow * Cc + n0 + ni0 * 16 + n]       = (__bf16)acc0[r];
    Out[(size_t)mrow * Cc + n0 + (ni0 + 1) * 16 + n] = (__bf16)acc1[r];
  }
}

// ---------------------------------------------------------------------------
// Kernel 2: fused causal attention for one (b, h, 64-query block).
// Grid: (T/64, H, B). Block: 256 threads = 8 waves. Dynamic LDS ~178KB.
// ---------------------------------------------------------------------------
__launch_bounds__(256, 1)
__global__ void attn_kernel(const __bf16* __restrict__ q,
                            const __bf16* __restrict__ k,
                            const __bf16* __restrict__ v,
                            float* __restrict__ out) {
  extern __shared__ char smem_raw[];
  __bf16* Ql  = (__bf16*)smem_raw;                 // [64][QL_LD]
  __bf16* Kl  = Ql + 64 * QL_LD;                   // [256][KL_LD]
  __bf16* Vt  = Kl + 256 * KL_LD;                  // [Dd][VT_LD]  V transposed
  __bf16* Pl  = Vt + (size_t)Dd * VT_LD;           // [64][PL_LD]  exp(S-max), bf16
  float*  Sl  = (float*)(Pl + 64 * PL_LD);         // [64][SL_LD]  raw scores f32
  float*  red = Sl + 64 * SL_LD;                   // [64][8]: rowmax(0..3) rowsum(4..7)

  const int tid  = threadIdx.x;
  const int lane = tid & 31;
  const int w    = tid >> 5;
  const int qb   = blockIdx.x;   // query block (0..3)
  const int h    = blockIdx.y;   // head
  const int b    = blockIdx.z;   // batch
  const size_t tok0 = (size_t)b * Tt;
  const int col0 = h * Dd;

  // --- stage Q (64x64) and K (256x64) via async b128 copies ---
#pragma unroll
  for (int i = 0; i < 2; ++i) {              // Q: 512 chunks of 16B
    int idx = tid + i * 256;
    int r = idx >> 3, co = (idx & 7) * 8;
    async_copy16(q + (tok0 + qb * 64 + r) * Cc + col0 + co, Ql + r * QL_LD + co);
  }
#pragma unroll
  for (int i = 0; i < 8; ++i) {              // K: 2048 chunks of 16B
    int idx = tid + i * 256;
    int r = idx >> 3, co = (idx & 7) * 8;
    async_copy16(k + (tok0 + r) * Cc + col0 + co, Kl + r * KL_LD + co);
  }
  // --- stage V transposed: b128 global read, b16 LDS scatter ---
#pragma unroll
  for (int i = 0; i < 8; ++i) {
    int idx = tid + i * 256;
    int r = idx >> 3, c0 = (idx & 7) * 8;
    union { uint4 q4; __bf16 h[8]; } vv;
    vv.q4 = *(const uint4*)(v + (tok0 + r) * Cc + col0 + c0);
#pragma unroll
    for (int j = 0; j < 8; ++j) Vt[(c0 + j) * VT_LD + r] = vv.h[j];
  }
  async_copy_fence();
  __syncthreads();

  // --- Phase 1: S = (1/sqrt(C)) * Q @ K^T, causal-masked, f32 to LDS ---
  {
    const float scale = 0.05103103630798288f;  // 1/sqrt(384)
    const int qi = w & 3;          // query 16-block within 64
    const int kh = w >> 2;         // key half (0/1): 8 key 16-blocks each
    v16bf a0 = load_a16(Ql + qi * 16 * QL_LD,      QL_LD, lane);
    v16bf a1 = load_a16(Ql + qi * 16 * QL_LD + 32, QL_LD, lane);
    const int n = lane & 15, mb = (lane >> 4) << 3;
#pragma unroll
    for (int kt = 0; kt < 8; ++kt) {
      int nb = kh * 8 + kt;
      v16bf b0 = load_bt16(Kl + nb * 16 * KL_LD,      KL_LD, lane);
      v16bf b1 = load_bt16(Kl + nb * 16 * KL_LD + 32, KL_LD, lane);
      v8f c = {0.f, 0.f, 0.f, 0.f, 0.f, 0.f, 0.f, 0.f};
      c = wmma_bf16(a0, b0, c);
      c = wmma_bf16(a1, b1, c);
      int gk = nb * 16 + n;
#pragma unroll
      for (int r = 0; r < 8; ++r) {
        int m  = qi * 16 + mb + r;
        int gq = qb * 64 + m;
        float s = c[r] * scale;
        if (gk > gq) s = -__builtin_inff();
        Sl[m * SL_LD + gk] = s;
      }
    }
  }
  __syncthreads();

  // --- Phase 2: rowmax + exp into bf16 P (unnormalized) + rowsum ---
  // 4 threads per row; normalization folded into the final output store.
  {
    const int row = tid >> 2, seg = tid & 3;
    const float* srow = Sl + row * SL_LD + seg * 64;
    __bf16* prow = Pl + row * PL_LD + seg * 64;
    float mx = -3.0e38f;
#pragma unroll 8
    for (int j = 0; j < 64; ++j) mx = fmaxf(mx, srow[j]);
    red[row * 8 + seg] = mx;
    __syncthreads();
    float rmx = fmaxf(fmaxf(red[row * 8 + 0], red[row * 8 + 1]),
                      fmaxf(red[row * 8 + 2], red[row * 8 + 3]));
    float sum = 0.f;
#pragma unroll 8
    for (int j = 0; j < 64; j += 2) {       // exp(-inf)=0 handles the mask
      float p0 = __expf(srow[j]     - rmx);
      float p1 = __expf(srow[j + 1] - rmx);
      sum += p0 + p1;
      union { __bf16 h[2]; uint32_t u; } pk;
      pk.h[0] = (__bf16)p0;
      pk.h[1] = (__bf16)p1;
      *(uint32_t*)(prow + j) = pk.u;        // packed ds_store_b32
    }
    red[row * 8 + 4 + seg] = sum;
  }
  __syncthreads();

  // --- Phase 3: O = P @ V, scaled by 1/rowsum at the store ---
  {
    const int mi  = w & 3;
    const int nb0 = (w >> 2) << 1;
    v8f o0 = {0.f, 0.f, 0.f, 0.f, 0.f, 0.f, 0.f, 0.f};
    v8f o1 = o0;
#pragma unroll
    for (int ks = 0; ks < 8; ++ks) {   // key dim in chunks of 32
      v16bf a  = load_a16(Pl + mi * 16 * PL_LD + ks * 32, PL_LD, lane);
      v16bf b0 = load_bt16(Vt + nb0 * 16 * VT_LD + ks * 32,       VT_LD, lane);
      v16bf b1 = load_bt16(Vt + (nb0 + 1) * 16 * VT_LD + ks * 32, VT_LD, lane);
      o0 = wmma_bf16(a, b0, o0);
      o1 = wmma_bf16(a, b1, o1);
    }
    const int n = lane & 15, mb = (lane >> 4) << 3;
#pragma unroll
    for (int r = 0; r < 8; ++r) {
      int m = mi * 16 + mb + r;
      float rs = red[m * 8 + 4] + red[m * 8 + 5] +
                 red[m * 8 + 6] + red[m * 8 + 7];
      float inv = 1.f / rs;
      size_t rowoff = (tok0 + qb * 64 + m) * Cc + col0;
      out[rowoff + nb0 * 16 + n]       = o0[r] * inv;
      out[rowoff + (nb0 + 1) * 16 + n] = o1[r] * inv;
    }
  }
}

// ---------------------------------------------------------------------------
// Host launcher
// ---------------------------------------------------------------------------
extern "C" void kernel_launch(void* const* d_in, const int* in_sizes, int n_in,
                              void* d_out, int out_size, void* d_ws, size_t ws_size,
                              hipStream_t stream) {
  (void)in_sizes; (void)n_in; (void)out_size; (void)ws_size;
  const float* x  = (const float*)d_in[0];
  const float* Wq = (const float*)d_in[1];
  const float* Wk = (const float*)d_in[2];
  const float* Wv = (const float*)d_in[3];
  float* out = (float*)d_out;

  const size_t nx = (size_t)BT * Cc;   // 12,582,912
  const size_t nw = (size_t)Cc * Cc;   //    147,456

  __bf16* xb  = (__bf16*)d_ws;
  __bf16* wqb = xb  + nx;
  __bf16* wkb = wqb + nw;
  __bf16* wvb = wkb + nw;
  __bf16* qb_ = wvb + nw;
  __bf16* kb_ = qb_ + nx;
  __bf16* vb_ = kb_ + nx;   // total ws: ~101 MB of bf16

  cvt_bf16_kernel<<<dim3((unsigned)(nx / 4 / 256)), dim3(256), 0, stream>>>(
      (const float4*)x, (uint2*)xb, (int)(nx / 4));
  cvt_bf16_kernel<<<dim3((unsigned)((nw / 4 + 255) / 256)), dim3(256), 0, stream>>>(
      (const float4*)Wq, (uint2*)wqb, (int)(nw / 4));
  cvt_bf16_kernel<<<dim3((unsigned)((nw / 4 + 255) / 256)), dim3(256), 0, stream>>>(
      (const float4*)Wk, (uint2*)wkb, (int)(nw / 4));
  cvt_bf16_kernel<<<dim3((unsigned)((nw / 4 + 255) / 256)), dim3(256), 0, stream>>>(
      (const float4*)Wv, (uint2*)wvb, (int)(nw / 4));

  qkv_gemm_kernel<<<dim3(BT / 64, Cc / 64, 3), dim3(256), 0, stream>>>(
      xb, wqb, wkb, wvb, qb_, kb_, vb_);

  const size_t smem =
      (size_t)(64 * QL_LD + 256 * KL_LD + Dd * VT_LD + 64 * PL_LD) * sizeof(__bf16) +
      (size_t)(64 * SL_LD + 64 * 8) * sizeof(float);   // ~178 KB

  attn_kernel<<<dim3(Tt / 64, Hh, Bb), dim3(256), smem, stream>>>(
      qb_, kb_, vb_, out);
}
